// CrfLayer_61443802137187
// MI455X (gfx1250) — compile-verified, matrix operand order checked by reading, not run on hardware
//
#include <hip/hip_runtime.h>
#include <cstdint>
#include <cstddef>

#define KT 128   // num tags
#define TT 512   // seq len
#define BB 512   // batch
#define MT 16    // batch rows per workgroup (WMMA M tile)

typedef __attribute__((ext_vector_type(16))) __bf16 v16bf;
typedef __attribute__((ext_vector_type(16))) unsigned short v16us;
typedef __attribute__((ext_vector_type(8))) float v8f;
typedef __attribute__((address_space(3))) void lds_void;

__device__ __forceinline__ unsigned short f2bf_rne(float x) {
  unsigned int u = __float_as_uint(x);
  u += 0x7fffu + ((u >> 16) & 1u);
  return (unsigned short)(u >> 16);
}
__device__ __forceinline__ float bf2f(unsigned short h) {
  return __uint_as_float(((unsigned int)h) << 16);
}

// ---------------------------------------------------------------------------
// Kernel 1: E = exp(trans) split into bf16 hi/lo; also copy trans to output.
// ---------------------------------------------------------------------------
__global__ void crf_prep(const float* __restrict__ trans,
                         float* __restrict__ out_trans,
                         unsigned short* __restrict__ Ehi,
                         unsigned short* __restrict__ Elo) {
  const int i = blockIdx.x * 256 + threadIdx.x;  // grid covers KT*KT
  const float tv = trans[i];
  out_trans[i] = tv;
  const float e = __expf(tv);
  const unsigned short hb = f2bf_rne(e);
  Ehi[i] = hb;
  Elo[i] = f2bf_rne(e - bf2f(hb));
}

// ---------------------------------------------------------------------------
// Kernel 2: gold-path sequence score (unary + binary), one block per batch b.
// ---------------------------------------------------------------------------
__global__ __launch_bounds__(128)
void crf_score(const float* __restrict__ em, const int* __restrict__ tags,
               const int* __restrict__ mask, const float* __restrict__ trans,
               float* __restrict__ score_ws) {
  const int b = blockIdx.x;
  const int tid = threadIdx.x;
  const float* emb = em + (size_t)b * TT * KT;
  const int* tg = tags + b * TT;
  const int* mk = mask + b * TT;
  float s = 0.f;
  for (int t = tid; t < TT; t += 128) {
    const float m = (float)mk[t];
    s += m * emb[(size_t)t * KT + tg[t]];
    if (t >= 1) s += m * trans[tg[t - 1] * KT + tg[t]];
  }
  __shared__ float red[128];
  red[tid] = s;
  __syncthreads();
  for (int off = 64; off > 0; off >>= 1) {
    if (tid < off) red[tid] += red[tid + off];
    __syncthreads();
  }
  if (tid == 0) score_ws[b] = red[0];
}

// ---------------------------------------------------------------------------
// Kernel 3: fused forward (logsumexp via WMMA) + Viterbi (max-plus, VALU).
// 256 threads: waves 0-3 = WMMA logsumexp, waves 4-7 = Viterbi + backptrs.
// Next-step emission tile is streamed with GLOBAL_LOAD_ASYNC_TO_LDS_B128
// (ASYNCcnt), drained just before the end-of-step barrier.
// ---------------------------------------------------------------------------
__global__ __launch_bounds__(256, 1)
void crf_fwd_viterbi(const float* __restrict__ em, const int* __restrict__ mask,
                     const float* __restrict__ trans,
                     const unsigned short* __restrict__ Ehi,
                     const unsigned short* __restrict__ Elo,
                     const float* __restrict__ score_ws,
                     unsigned char* __restrict__ bp_ws,
                     int* __restrict__ lasttag_ws,
                     float* __restrict__ out_ll,
                     float* __restrict__ out_pred) {
  __shared__ float trans_s[KT * KT];     // 64KB, viterbi max-plus operand
  __shared__ float a_l[2][MT * KT];      // logsumexp alpha, ping-pong
  __shared__ float a_v[2][MT * KT];      // viterbi alpha, ping-pong
  __shared__ float em_s[2][MT * KT];     // emission tile, ping-pong
  __shared__ float mpart[4][MT];         // rowmax partials from 4 GEMM waves
  __shared__ float m_s[MT];              // rowmax of alpha_lse (stability)
  __shared__ int mk_s[2][MT];            // mask column, ping-pong

  const int tid = threadIdx.x;
  const int wave = tid >> 5;
  const int lane = tid & 31;
  const int b0 = blockIdx.x * MT;

  // transition matrix -> LDS (viterbi)
  for (int q = tid; q < KT * KT; q += 256) trans_s[q] = trans[q];

  // alpha0 = em[:,0,:]; preload em[:,1,:] and mask[:,1]
  for (int q = tid; q < MT * KT; q += 256) {
    const int r = q >> 7, c = q & 127;
    const float v0 = em[((size_t)(b0 + r) * TT + 0) * KT + c];
    a_l[0][q] = v0;
    a_v[0][q] = v0;
    em_s[0][q] = em[((size_t)(b0 + r) * TT + 1) * KT + c];
  }
  if (tid < MT) mk_s[0][tid] = mask[(b0 + tid) * TT + 1];
  __syncthreads();

  // initial rowmax of alpha0
  if (tid < 128) {
    const int r = tid >> 3, seg = tid & 7;
    float mx = -3.0e38f;
    for (int c = seg * 16; c < seg * 16 + 16; ++c) mx = fmaxf(mx, a_l[0][r * KT + c]);
    mx = fmaxf(mx, __shfl_xor(mx, 1, 8));
    mx = fmaxf(mx, __shfl_xor(mx, 2, 8));
    mx = fmaxf(mx, __shfl_xor(mx, 4, 8));
    if (seg == 0) m_s[r] = mx;
  }

  // WMMA fragment lane coordinates (16-bit A/B layout, wave32)
  const int hi16 = (lane >> 4) & 1;
  const int lr = lane & 15;
  const int nt0 = wave * 2;  // GEMM waves own column tiles nt0, nt0+1

  // E fragments (bf16 hi/lo) resident in VGPRs for the GEMM waves
  v16us ehi[2][4], elo[2][4];
  if (wave < 4) {
#pragma unroll
    for (int nt = 0; nt < 2; ++nt) {
      const int n = (nt0 + nt) * 16 + lr;
#pragma unroll
      for (int kc = 0; kc < 4; ++kc) {
#pragma unroll
        for (int e = 0; e < 16; ++e) {
          const int k = kc * 32 + ((e < 8) ? 0 : 16) + (hi16 ? 8 : 0) + (e & 7);
          ehi[nt][kc][e] = Ehi[k * KT + n];
          elo[nt][kc][e] = Elo[k * KT + n];
        }
      }
    }
  }
  __syncthreads();  // m_s + LDS state ready

  // each thread owns 32 contiguous bytes of the 8KB emission tile
  const int ar = tid >> 4;           // row 0..15
  const int ac = (tid & 15) * 8;     // col (8 floats)

  for (int t = 1; t < TT; ++t) {
    const int cur = (t - 1) & 1;
    const int nxt = t & 1;

    // async-copy next step's emission tile straight into LDS (ASYNCcnt)
    int mkr = 0;
    if (t + 1 < TT) {
      const float* gsrc = &em[((size_t)(b0 + ar) * TT + (t + 1)) * KT + ac];
      const unsigned int ldst =
          (unsigned int)(size_t)(lds_void*)(void*)&em_s[nxt][tid * 8];
      asm volatile("global_load_async_to_lds_b128 %0, %1, off"
                   :: "v"(ldst), "v"(gsrc) : "memory");
      asm volatile("global_load_async_to_lds_b128 %0, %1, off offset:16"
                   :: "v"(ldst), "v"(gsrc) : "memory");
      if (tid < MT) mkr = mask[(b0 + tid) * TT + (t + 1)];
    }

    // combine rowmax partials from previous step
    if (t > 1 && tid < MT) {
      m_s[tid] = fmaxf(fmaxf(mpart[0][tid], mpart[1][tid]),
                       fmaxf(mpart[2][tid], mpart[3][tid]));
    }
    __syncthreads();

    if (wave < 4) {
      // ---------------- logsumexp via WMMA: S = P * E ----------------
      const float mrow = m_s[lr];
      v8f c0 = {};
      v8f c1 = {};
#pragma unroll
      for (int kc = 0; kc < 4; ++kc) {
        v16bf ph, pl;
#pragma unroll
        for (int e = 0; e < 16; ++e) {
          const int k = kc * 32 + ((e < 8) ? 0 : 16) + (hi16 ? 8 : 0) + (e & 7);
          const float p = __expf(a_l[cur][lr * KT + k] - mrow);
          const __bf16 hb = (__bf16)p;            // hw cvt, RNE
          ph[e] = hb;
          pl[e] = (__bf16)(p - (float)hb);        // residual
        }
        const v16bf e0h = __builtin_bit_cast(v16bf, ehi[0][kc]);
        const v16bf e0l = __builtin_bit_cast(v16bf, elo[0][kc]);
        const v16bf e1h = __builtin_bit_cast(v16bf, ehi[1][kc]);
        const v16bf e1l = __builtin_bit_cast(v16bf, elo[1][kc]);
        // alternate independent accumulator chains to hide WMMA RAW hazards
        c0 = __builtin_amdgcn_wmma_f32_16x16x32_bf16(false, ph, false, e0h, (short)0, c0, false, false);
        c1 = __builtin_amdgcn_wmma_f32_16x16x32_bf16(false, ph, false, e1h, (short)0, c1, false, false);
        c0 = __builtin_amdgcn_wmma_f32_16x16x32_bf16(false, ph, false, e0l, (short)0, c0, false, false);
        c1 = __builtin_amdgcn_wmma_f32_16x16x32_bf16(false, ph, false, e1l, (short)0, c1, false, false);
        c0 = __builtin_amdgcn_wmma_f32_16x16x32_bf16(false, pl, false, e0h, (short)0, c0, false, false);
        c1 = __builtin_amdgcn_wmma_f32_16x16x32_bf16(false, pl, false, e1h, (short)0, c1, false, false);
      }
      // alpha' = log(S) + m + em (masked), and rowmax for next step
#pragma unroll
      for (int v = 0; v < 8; ++v) {
        const int Mv = hi16 ? (v + 8) : v;
        const int mk = mk_s[cur][Mv];
        const float madd = m_s[Mv];
        float r0v, r1v;
        {
          const int col = nt0 * 16 + lr;
          const float nw = __logf(c0[v]) + madd + em_s[cur][Mv * KT + col];
          const float od = a_l[cur][Mv * KT + col];
          r0v = mk ? nw : od;
          a_l[nxt][Mv * KT + col] = r0v;
        }
        {
          const int col = (nt0 + 1) * 16 + lr;
          const float nw = __logf(c1[v]) + madd + em_s[cur][Mv * KT + col];
          const float od = a_l[cur][Mv * KT + col];
          r1v = mk ? nw : od;
          a_l[nxt][Mv * KT + col] = r1v;
        }
        float mx = fmaxf(r0v, r1v);
        mx = fmaxf(mx, __shfl_xor(mx, 1, 16));
        mx = fmaxf(mx, __shfl_xor(mx, 2, 16));
        mx = fmaxf(mx, __shfl_xor(mx, 4, 16));
        mx = fmaxf(mx, __shfl_xor(mx, 8, 16));
        if (lr == 0) mpart[wave][Mv] = mx;
      }
    } else {
      // ---------------- Viterbi max-plus + backpointers ----------------
      const int vt = tid - 128;
      const int r0 = (vt >> 5) * 4;  // 4 rows
      const int j0 = (vt & 31) * 4;  // 4 cols
      float best[16];
      int barg[16];
#pragma unroll
      for (int q = 0; q < 16; ++q) { best[q] = -3.0e38f; barg[q] = 0; }
#pragma unroll 2
      for (int i = 0; i < KT; ++i) {
        const float4 tv = *(const float4*)&trans_s[i * KT + j0];
        const float tvv[4] = {tv.x, tv.y, tv.z, tv.w};
#pragma unroll
        for (int rr = 0; rr < 4; ++rr) {
          const float av = a_v[cur][(r0 + rr) * KT + i];
#pragma unroll
          for (int jj = 0; jj < 4; ++jj) {
            const float cand = av + tvv[jj];
            const int q = rr * 4 + jj;
            if (cand > best[q]) { best[q] = cand; barg[q] = i; }
          }
        }
      }
#pragma unroll
      for (int rr = 0; rr < 4; ++rr) {
        const int r = r0 + rr;
        const int mk = mk_s[cur][r];
        unsigned int packed = 0;
#pragma unroll
        for (int jj = 0; jj < 4; ++jj) {
          const int j = j0 + jj;
          const int q = rr * 4 + jj;
          const float nw = best[q] + em_s[cur][r * KT + j];
          const float od = a_v[cur][r * KT + j];
          a_v[nxt][r * KT + j] = mk ? nw : od;
          const int bpv = mk ? barg[q] : j;  // identity bp when masked
          packed |= ((unsigned int)bpv & 0xffu) << (8 * jj);
        }
        *(unsigned int*)(bp_ws + ((size_t)(t - 1) * BB + (b0 + r)) * KT + j0) = packed;
      }
    }

    // stage next mask column; drain async emission copy, then step barrier
    if (t + 1 < TT) {
      if (tid < MT) mk_s[nxt][tid] = mkr;
      asm volatile("s_wait_asynccnt 0x0" ::: "memory");
    }
    __syncthreads();
  }

  // epilogue: logZ, log-likelihood, last viterbi tag
  const int fin = (TT - 1) & 1;
  if (tid < MT) {
    const int r = tid, b = b0 + tid;
    float mx = -3.0e38f;
    for (int c = 0; c < KT; ++c) mx = fmaxf(mx, a_l[fin][r * KT + c]);
    float s = 0.f;
    for (int c = 0; c < KT; ++c) s += __expf(a_l[fin][r * KT + c] - mx);
    out_ll[b] = score_ws[b] - (__logf(s) + mx);
    float bmx = -3.0e38f;
    int bi = 0;
    for (int c = 0; c < KT; ++c) {
      const float v = a_v[fin][r * KT + c];
      if (v > bmx) { bmx = v; bi = c; }
    }
    lasttag_ws[b] = bi;
    out_pred[(size_t)b * TT + (TT - 1)] = (float)bi;
  }
}

// ---------------------------------------------------------------------------
// Kernel 4: backtrace. 8-deep software pipeline of backpointer rows into an
// LDS ring so the dependent chase hits LDS latency, not HBM/L2 latency.
// 128 threads handle 32 batch elements.
// ---------------------------------------------------------------------------
__global__ __launch_bounds__(128, 1)
void crf_backtrace(const unsigned char* __restrict__ bp_ws,
                   const int* __restrict__ lasttag_ws,
                   float* __restrict__ out_pred) {
  __shared__ unsigned char rb[8][32 * KT];  // 32KB ring
  const int tid = threadIdx.x;
  const int b0 = blockIdx.x * 32;
  const int row = tid >> 2;  // 0..31: local batch row
  const int seg = tid & 3;   // 4 threads x 32B cover one 128B row

  int carry = (tid < 32) ? lasttag_ws[b0 + tid] : 0;

  uint4 pv[8][2];
#pragma unroll
  for (int d = 0; d < 8; ++d) {
    const int k = (TT - 2) - d;
    const uint4* s = (const uint4*)(bp_ws + ((size_t)k * BB + b0 + row) * KT + seg * 32);
    pv[d][0] = s[0];
    pv[d][1] = s[1];
  }

  for (int base = TT - 2; base >= 7; base -= 8) {
#pragma unroll
    for (int d = 0; d < 8; ++d) {
      const int k = base - d;
      uint4* dst = (uint4*)(rb[d] + row * KT + seg * 32);
      dst[0] = pv[d][0];
      dst[1] = pv[d][1];
      __syncthreads();
      if (tid < 32) {
        const int prev = rb[d][tid * KT + carry];
        out_pred[(size_t)(b0 + tid) * TT + k] = (float)prev;
        carry = prev;
      }
      const int kn = k - 8;
      if (kn >= 0) {
        const uint4* s = (const uint4*)(bp_ws + ((size_t)kn * BB + b0 + row) * KT + seg * 32);
        pv[d][0] = s[0];
        pv[d][1] = s[1];
      }
      __syncthreads();
    }
  }
  // remainder k = 6..0: short dependent chain straight from global
  if (tid < 32) {
    for (int k = 6; k >= 0; --k) {
      const int prev = bp_ws[((size_t)k * BB + b0 + tid) * KT + carry];
      out_pred[(size_t)(b0 + tid) * TT + k] = (float)prev;
      carry = prev;
    }
  }
}

// ---------------------------------------------------------------------------
// Host launcher
// ---------------------------------------------------------------------------
extern "C" void kernel_launch(void* const* d_in, const int* in_sizes, int n_in,
                              void* d_out, int out_size, void* d_ws, size_t ws_size,
                              hipStream_t stream) {
  (void)in_sizes; (void)n_in; (void)out_size; (void)ws_size;
  const float* em    = (const float*)d_in[0];  // (B,T,K) f32
  const int* tags    = (const int*)d_in[1];    // (B,T) i32
  const int* mask    = (const int*)d_in[2];    // (B,T) i32
  const float* trans = (const float*)d_in[3];  // (K,K) f32

  float* out = (float*)d_out;
  float* out_ll    = out;                   // B
  float* out_trans = out + BB;              // K*K
  float* out_pred  = out + BB + KT * KT;    // B*T (tags as floats)

  unsigned char* ws = (unsigned char*)d_ws;
  unsigned char* bp_ws = ws;  // (T-1)*B*K bytes = 33,488,896
  const size_t off = 33554432;  // 32 MiB aligned region for small buffers
  float* score_ws = (float*)(ws + off);                 // B floats
  int* lt_ws      = (int*)(ws + off + 4096);            // B ints
  unsigned short* Ehi = (unsigned short*)(ws + off + 8192);   // K*K bf16
  unsigned short* Elo = Ehi + KT * KT;                        // K*K bf16

  crf_prep<<<(KT * KT) / 256, 256, 0, stream>>>(trans, out_trans, Ehi, Elo);
  crf_score<<<BB, 128, 0, stream>>>(em, tags, mask, trans, score_ws);
  crf_fwd_viterbi<<<BB / MT, 256, 0, stream>>>(em, mask, trans, Ehi, Elo,
                                               score_ws, bp_ws, lt_ws,
                                               out_ll, out_pred);
  crf_backtrace<<<BB / 32, 128, 0, stream>>>(bp_ws, lt_ws, out_pred);
}
